// QEffDeepseekV3MoE_83700322664924
// MI455X (gfx1250) — matrix operand order repaired and checked by reading, not compiled, not run on hardware
//
#include <hip/hip_runtime.h>
#include <hip/hip_bf16.h>

// DeepSeek-V3 quantized MoE for MI455X (gfx1250, wave32, WMMA).
// E=64, D=2048, I=1536, G=128, S=8, K=4.
// HBM-bound (~150MB int4 weights): stream weights once, dequant (q-z) into
// exact integer-valued f16 with packed VALU (0x6400 exponent trick), MAC on the
// matrix pipe with per-group C=0 restarts, fold group scale in with f32 FMAs
// afterwards (exactly like the reference's fp32 math, and kills all pk_muls).
// Two independent WMMA chains per matrix remove D->C hazard bubbles.

typedef __attribute__((ext_vector_type(16))) _Float16 v16h;
typedef __attribute__((ext_vector_type(8)))  _Float16 v8h;
typedef __attribute__((ext_vector_type(2)))  _Float16 v2h;
typedef __attribute__((ext_vector_type(8)))  float    v8f;

#define E_DIM 64
#define D_DIM 2048
#define I_DIM 1536
#define GQ    128
#define S_DIM 8
#define K_TOP 4
#define T_DIM (S_DIM * K_TOP)

// 8 nibbles (packed dword, low nibble first) -> 8 f16 holding exact (q - z).
// f16 bits (0x6400 | n) == 1024 + n exactly; zo holds packed f16(1024 + z).
static __device__ inline v8h dq8(unsigned q, v2h zo) {
  v8h r;
#pragma unroll
  for (int p = 0; p < 4; ++p) {
    unsigned packed = 0x64006400u | (q & 0xFu) | ((q & 0xF0u) << 12);
    v2h w = __builtin_bit_cast(v2h, packed) - zo;  // exact: (1024+n)-(1024+z)
    r[2 * p]     = w[0];
    r[2 * p + 1] = w[1];
    q >>= 8;
  }
  return r;
}

// Packed f16(1024+z) pair straight from the zero-point nibble: 0x6400 | z.
static __device__ inline v2h zpack(unsigned zn) {
  return __builtin_bit_cast(v2h, 0x64006400u | zn | (zn << 16));
}

static __device__ inline v16h cat16(v8h a, v8h b) {
  return __builtin_shufflevector(a, b, 0, 1, 2, 3, 4, 5, 6, 7,
                                       8, 9, 10, 11, 12, 13, 14, 15);
}

static __device__ inline v8f wmma_f16(v16h a, v16h b, v8f c) {
  return __builtin_amdgcn_wmma_f32_16x16x32_f16(false, a, false, b,
                                                (short)0, c, false, false);
}

// ---------------- Phase 1: h[t,:] = silu(W_gate x) * (W_up x) ----------------
// grid = (T, I/64), block = 128 (4 waves). Wave w owns 16-row tile 4*blockIdx.y+w;
// the 4 waves share one LDS copy of x[s] (f16).
// A-layout (f16 A 16x32): lane<16 -> row M=lane, K{0..7,16..23}; lane>=16 -> K{8..15,24..31}.
// B: every lane loads the same x-chunk -> all C columns identical; no divergence.
__global__ __launch_bounds__(128) void moe_gateup_kernel(
    const float* __restrict__ x, const int* __restrict__ topk_idx,
    const unsigned char* __restrict__ gqw, const float* __restrict__ gsc,
    const unsigned char* __restrict__ gqz,
    const unsigned char* __restrict__ uqw, const float* __restrict__ usc,
    const unsigned char* __restrict__ uqz,
    _Float16* __restrict__ hbuf) {
  const int t    = (int)blockIdx.x;
  const int wave = (int)threadIdx.x >> 5;
  const int lane = (int)threadIdx.x & 31;
  const int tile = (int)blockIdx.y * 4 + wave;
  const int s    = t / K_TOP;
  const int e    = topk_idx[t];
  const int hi   = lane >> 4;

  __shared__ __align__(16) _Float16 xs[D_DIM];
  {
    const float4* xv = (const float4*)(x + (size_t)s * D_DIM);
    for (int i = (int)threadIdx.x; i < D_DIM / 4; i += 128) {
      float4 f = xv[i];
      xs[4 * i + 0] = (_Float16)f.x;
      xs[4 * i + 1] = (_Float16)f.y;
      xs[4 * i + 2] = (_Float16)f.z;
      xs[4 * i + 3] = (_Float16)f.w;
    }
  }
  __syncthreads();

  const int    row = tile * 16 + (lane & 15);
  const size_t rg  = (size_t)e * I_DIM + (size_t)row;
  const unsigned char* gw  = gqw + rg * (D_DIM / 2);
  const unsigned char* uw  = uqw + rg * (D_DIM / 2);
  const float*         gs  = gsc + rg * (D_DIM / GQ);
  const float*         us  = usc + rg * (D_DIM / GQ);
  const unsigned char* gz8 = gqz + rg * (D_DIM / (2 * GQ));
  const unsigned char* uz8 = uqz + rg * (D_DIM / (2 * GQ));

  v8f accg = {}, accu = {};

  for (int grp = 0; grp < D_DIM / GQ; ++grp) {
    const float sg = gs[grp];
    const float su = us[grp];
    const unsigned sh = (grp & 1) * 4;
    const v2h gzo = zpack((gz8[grp >> 1] >> sh) & 0xF);
    const v2h uzo = zpack((uz8[grp >> 1] >> sh) & 0xF);

    const unsigned char* gB = gw + grp * (GQ / 2);
    const unsigned char* uB = uw + grp * (GQ / 2);
    __builtin_prefetch(gB + 256, 0, 1);
    __builtin_prefetch(uB + 256, 0, 1);

    v8f g0 = {}, g1 = {}, u0 = {}, u1 = {};  // per-group C=0 restarts
#pragma unroll
    for (int c = 0; c < GQ / 32; ++c) {
      const int bo = c * 16 + hi * 4;
      const unsigned gq0 = *(const unsigned*)(gB + bo);
      const unsigned gq1 = *(const unsigned*)(gB + bo + 8);
      const unsigned uq0 = *(const unsigned*)(uB + bo);
      const unsigned uq1 = *(const unsigned*)(uB + bo + 8);

      const v16h ag = cat16(dq8(gq0, gzo), dq8(gq1, gzo));
      const v16h au = cat16(dq8(uq0, uzo), dq8(uq1, uzo));

      const v8h* bp = (const v8h*)&xs[grp * GQ + c * 32 + hi * 8];
      const v16h bv = cat16(bp[0], bp[2]);

      if (c & 1) { g1 = wmma_f16(ag, bv, g1); u1 = wmma_f16(au, bv, u1); }
      else       { g0 = wmma_f16(ag, bv, g0); u0 = wmma_f16(au, bv, u0); }
    }
#pragma unroll
    for (int j = 0; j < 8; ++j) {
      accg[j] += sg * (g0[j] + g1[j]);
      accu[j] += su * (u0[j] + u1[j]);
    }
  }

  // C layout: lane0 VGPR j -> M=j; lane16 VGPR j -> M=8+j. One writer per value.
  if ((lane & 15) == 0) {
    v8h hv;
#pragma unroll
    for (int j = 0; j < 8; ++j) {
      const float g = accg[j], u = accu[j];
      hv[j] = (_Float16)((g / (1.0f + __expf(-g))) * u);  // silu(g)*u
    }
    *(v8h*)(hbuf + (size_t)t * I_DIM + tile * 16 + hi * 8) = hv;
  }
}

// -------- Phase 2: out[s,:] = sum_k w[s,k] * (W_down[e_k] h[s*K+k]) --------
// grid = (S, D/16), block = 128 (4 waves): wave k handles expert slot k, then a
// small LDS reduction combines the K=4 partials. Deterministic, no atomics.
__global__ __launch_bounds__(128) void moe_down_kernel(
    const _Float16* __restrict__ hbuf, const int* __restrict__ topk_idx,
    const float* __restrict__ topk_w,
    const unsigned char* __restrict__ dqw, const float* __restrict__ dsc,
    const unsigned char* __restrict__ dqz,
    float* __restrict__ out) {
  const int s    = (int)blockIdx.x;
  const int tile = (int)blockIdx.y;
  const int wave = (int)threadIdx.x >> 5;
  const int lane = (int)threadIdx.x & 31;
  const int hi   = lane >> 4;
  const int row  = tile * 16 + (lane & 15);

  const int   t  = s * K_TOP + wave;
  const int   e  = topk_idx[t];
  const float wt = topk_w[t];

  __shared__ __align__(16) _Float16 hs[K_TOP][I_DIM];  // 12 KB
  __shared__ float red[K_TOP][16];
  {
    const v8h* hv = (const v8h*)(hbuf + (size_t)t * I_DIM);
    v8h*       hd = (v8h*)&hs[wave][0];
    for (int i = lane; i < I_DIM / 8; i += 32) hd[i] = hv[i];
  }
  __syncthreads();

  const size_t rg = (size_t)e * D_DIM + (size_t)row;
  const unsigned char* w  = dqw + rg * (I_DIM / 2);
  const float*         sc = dsc + rg * (I_DIM / GQ);
  const unsigned char* z8 = dqz + rg * (I_DIM / (2 * GQ));

  v8f acc = {};
  for (int grp = 0; grp < I_DIM / GQ; ++grp) {
    const float sf = sc[grp];
    const unsigned sh = (grp & 1) * 4;
    const v2h zo = zpack((z8[grp >> 1] >> sh) & 0xF);

    const unsigned char* wB = w + grp * (GQ / 2);
    __builtin_prefetch(wB + 256, 0, 1);

    v8f d0 = {}, d1 = {};
#pragma unroll
    for (int c = 0; c < GQ / 32; ++c) {
      const int bo = c * 16 + hi * 4;
      const unsigned w0 = *(const unsigned*)(wB + bo);
      const unsigned w1 = *(const unsigned*)(wB + bo + 8);
      const v16h a = cat16(dq8(w0, zo), dq8(w1, zo));

      const v8h* bp = (const v8h*)&hs[wave][grp * GQ + c * 32 + hi * 8];
      const v16h bv = cat16(bp[0], bp[2]);

      if (c & 1) d1 = wmma_f16(a, bv, d1);
      else       d0 = wmma_f16(a, bv, d0);
    }
#pragma unroll
    for (int j = 0; j < 8; ++j) acc[j] += sf * (d0[j] + d1[j]);
  }

  if ((lane & 15) == 0) {
#pragma unroll
    for (int j = 0; j < 8; ++j) red[wave][hi * 8 + j] = wt * acc[j];
  }
  __syncthreads();

  if (wave == 0 && (lane & 15) == 0) {
    float o[8];
#pragma unroll
    for (int j = 0; j < 8; ++j) {
      const int m = hi * 8 + j;
      o[j] = red[0][m] + red[1][m] + red[2][m] + red[3][m];
    }
    float4* orow = (float4*)(out + (size_t)s * D_DIM + tile * 16 + hi * 8);
    orow[0] = make_float4(o[0], o[1], o[2], o[3]);
    orow[1] = make_float4(o[4], o[5], o[6], o[7]);
  }
}

extern "C" void kernel_launch(void* const* d_in, const int* in_sizes, int n_in,
                              void* d_out, int out_size, void* d_ws, size_t ws_size,
                              hipStream_t stream) {
  (void)in_sizes; (void)n_in; (void)out_size; (void)ws_size;

  const float*         x    = (const float*)d_in[0];
  const int*           tidx = (const int*)d_in[1];
  const float*         tw   = (const float*)d_in[2];
  const unsigned char* gqw  = (const unsigned char*)d_in[3];
  const float*         gsc  = (const float*)d_in[4];
  const unsigned char* gqz  = (const unsigned char*)d_in[5];
  const unsigned char* uqw  = (const unsigned char*)d_in[6];
  const float*         usc  = (const float*)d_in[7];
  const unsigned char* uqz  = (const unsigned char*)d_in[8];
  const unsigned char* dqw  = (const unsigned char*)d_in[9];
  const float*         dsc  = (const float*)d_in[10];
  const unsigned char* dqz  = (const unsigned char*)d_in[11];
  // d_in[12] = group_size (hardcoded GQ=128)

  float*    out  = (float*)d_out;
  _Float16* hbuf = (_Float16*)d_ws;  // [T, I] f16 = 96 KB

  dim3 grid1(T_DIM, I_DIM / 64);  // 32 x 24 blocks, 4 waves each = 3072 waves
  moe_gateup_kernel<<<grid1, 128, 0, stream>>>(x, tidx, gqw, gsc, gqz,
                                               uqw, usc, uqz, hbuf);

  dim3 grid2(S_DIM, D_DIM / 16);  // 8 x 128 blocks, 4 waves each = 4096 waves
  moe_down_kernel<<<grid2, 128, 0, stream>>>(hbuf, tidx, tw, dqw, dsc, dqz, out);
}
